// BitBlock_78245714199349
// MI455X (gfx1250) — 16-bit and 8-bit matrix operand order repaired and checked by reading, compiled, not run on hardware
//
#include <hip/hip_runtime.h>
#include <stdint.h>

#define D_MODEL 1024
#define HIDDEN  4096
#define NTOK    4096
#define SEQ_T   2048
#define NHEADS  16
#define DHEAD   64

typedef __attribute__((ext_vector_type(8)))  int      v8i;
typedef __attribute__((ext_vector_type(8)))  float    v8f;
typedef __attribute__((ext_vector_type(16))) _Float16 v16h;
typedef __attribute__((ext_vector_type(4)))  unsigned int u32x4;
typedef __attribute__((ext_vector_type(4)))  int      i32x4;
typedef __attribute__((ext_vector_type(8)))  int      i32x8;

union U8I  { v8i  v; int   i32[8]; unsigned long long u64[4]; };
union U8F  { v8f  v; float f[8]; };
union U16H { v16h v; unsigned long long u64[4]; };
union HU   { _Float16 h; unsigned short u; };

#if defined(__gfx1250__) && __has_builtin(__builtin_amdgcn_tensor_load_to_lds) && \
    __has_builtin(__builtin_amdgcn_s_wait_tensorcnt)
#define USE_TDM 1
#else
#define USE_TDM 0
#endif

#if USE_TDM
__device__ __forceinline__ void tdm_load_2d(const void* gptr, unsigned lds_off,
                                            unsigned tile_w, unsigned tile_h,
                                            unsigned stride,
                                            unsigned tensor_w, unsigned tensor_h) {
  const unsigned long long ga = (unsigned long long)(size_t)gptr;
  u32x4 g0;
  g0[0] = 1u;
  g0[1] = lds_off;
  g0[2] = (unsigned)(ga & 0xFFFFFFFFu);
  g0[3] = (unsigned)((ga >> 32) & 0x1FFFFFFu) | (2u << 30);
  i32x8 g1;
  g1[0] = 0;
  g1[1] = (int)((tensor_w & 0xFFFFu) << 16);
  g1[2] = (int)(((tensor_w >> 16) & 0xFFFFu) | ((tensor_h & 0xFFFFu) << 16));
  g1[3] = (int)(((tensor_h >> 16) & 0xFFFFu) | ((tile_w & 0xFFFFu) << 16));
  g1[4] = (int)(tile_h & 0xFFFFu);
  g1[5] = (int)stride;
  g1[6] = 0;
  g1[7] = 0;
  const i32x4 z4 = {0, 0, 0, 0};
  const i32x8 z8 = {0, 0, 0, 0, 0, 0, 0, 0};
  __builtin_amdgcn_tensor_load_to_lds(g0, g1, z4, z4, z8, 0);
}
#endif

__device__ __forceinline__ float blk_sum(float v, float* red) {
  const int tid = threadIdx.x;
  red[tid] = v; __syncthreads();
  for (int off = 128; off > 0; off >>= 1) {
    if (tid < off) red[tid] += red[tid + off];
    __syncthreads();
  }
  float r = red[0]; __syncthreads();
  return r;
}
__device__ __forceinline__ float blk_max(float v, float* red) {
  const int tid = threadIdx.x;
  red[tid] = v; __syncthreads();
  for (int off = 128; off > 0; off >>= 1) {
    if (tid < off) red[tid] = fmaxf(red[tid], red[tid + off]);
    __syncthreads();
  }
  float r = red[0]; __syncthreads();
  return r;
}

__global__ __launch_bounds__(256) void k_abs_part(const float* __restrict__ w, int n,
                                                  float* __restrict__ partial) {
  __shared__ float red[256];
  float s = 0.f;
  for (int i = blockIdx.x * 256 + threadIdx.x; i < n; i += 256 * gridDim.x)
    s += fabsf(w[i]);
  float t = blk_sum(s, red);
  if (threadIdx.x == 0) partial[blockIdx.x] = t;
}

__global__ __launch_bounds__(256) void k_abs_fin(const float* __restrict__ partial, int n,
                                                 float* __restrict__ gw) {
  __shared__ float red[256];
  float t = blk_sum(partial[threadIdx.x], red);
  if (threadIdx.x == 0) gw[0] = fmaxf(t / (float)n, 1e-5f);
}

__global__ __launch_bounds__(256) void k_quant_w_t(const float* __restrict__ w,
                                                   const float* __restrict__ gwp,
                                                   signed char* __restrict__ wt,
                                                   int N, int K) {
  const float g = gwp[0];
  const int n_el = N * K;
  for (int i = blockIdx.x * 256 + threadIdx.x; i < n_el; i += 256 * gridDim.x) {
    float t = rintf(w[i] / g);
    t = fminf(fmaxf(t, -1.f), 1.f);
    const int n = i / K, k = i % K;
    wt[(size_t)k * N + n] = (signed char)t;
  }
}

__global__ __launch_bounds__(256) void k_ln_quant(const float* __restrict__ x,
                                                  const float* __restrict__ g,
                                                  const float* __restrict__ b,
                                                  signed char* __restrict__ xq,
                                                  float* __restrict__ xs) {
  __shared__ float red[256];
  const int row = blockIdx.x, tid = threadIdx.x;
  const float* xr = x + (size_t)row * D_MODEL;
  float v[4]; float s = 0.f;
#pragma unroll
  for (int i = 0; i < 4; i++) { v[i] = xr[tid + 256 * i]; s += v[i]; }
  const float mu = blk_sum(s, red) * (1.f / D_MODEL);
  s = 0.f;
#pragma unroll
  for (int i = 0; i < 4; i++) { float d = v[i] - mu; s += d * d; }
  const float var = blk_sum(s, red) * (1.f / D_MODEL);
  const float inv = rsqrtf(var + 1e-5f);
  float y[4]; float am = 0.f;
#pragma unroll
  for (int i = 0; i < 4; i++) {
    const int c = tid + 256 * i;
    y[i] = (v[i] - mu) * inv * g[c] + b[c];
    am = fmaxf(am, fabsf(y[i]));
  }
  const float gamma = fmaxf(blk_max(am, red), 1e-5f);
  const float sc = 127.f / gamma;
#pragma unroll
  for (int i = 0; i < 4; i++) {
    float t = rintf(y[i] * sc);
    t = fminf(fmaxf(t, -128.f), 127.f);
    xq[(size_t)row * D_MODEL + tid + 256 * i] = (signed char)t;
  }
  if (tid == 0) xs[row] = gamma * (1.f / 127.f);
}

__global__ __launch_bounds__(256) void k_quant_rows(const float* __restrict__ x,
                                                    signed char* __restrict__ xq,
                                                    float* __restrict__ xs, int C) {
  __shared__ float red[256];
  const int row = blockIdx.x, tid = threadIdx.x;
  const float* xr = x + (size_t)row * C;
  float am = 0.f;
  for (int i = tid; i < C; i += 256) am = fmaxf(am, fabsf(xr[i]));
  const float gamma = fmaxf(blk_max(am, red), 1e-5f);
  const float sc = 127.f / gamma;
  for (int i = tid; i < C; i += 256) {
    float t = rintf(xr[i] * sc);
    t = fminf(fmaxf(t, -128.f), 127.f);
    xq[(size_t)row * C + i] = (signed char)t;
  }
  if (tid == 0) xs[row] = gamma * (1.f / 127.f);
}

template <int MODE>
__global__ __launch_bounds__(128) void k_gemm_iu8(
    const signed char* __restrict__ A, const signed char* __restrict__ Wt,
    const float* __restrict__ xs, const float* __restrict__ gwp,
    const float* __restrict__ bias, const float* __restrict__ resid,
    float* __restrict__ outf, unsigned short* __restrict__ outh,
    const float* __restrict__ gate,
    int M, int N, int K) {
  __shared__ __align__(16) signed char sA[64 * 64];
  __shared__ __align__(16) signed char sB[64 * 64];
  const int tid = threadIdx.x;
  const int wave = tid >> 5, lane = tid & 31;
  const int lhi = lane >> 4, ln = lane & 15;
  const int bm = blockIdx.y * 64, bn = blockIdx.x * 64;

  U8I acc[4];
#pragma unroll
  for (int t = 0; t < 4; t++)
#pragma unroll
    for (int e = 0; e < 8; e++) acc[t].i32[e] = 0;

#if USE_TDM
  const unsigned ldsA = (unsigned)(size_t)(void*)sA;
  const unsigned ldsB = (unsigned)(size_t)(void*)sB;
#else
  const int lr = tid >> 2;
  const int lc = (tid & 3) * 16;
#endif

  for (int k0 = 0; k0 < K; k0 += 64) {
#if USE_TDM
    if (wave == 0) {
      tdm_load_2d(A + (size_t)bm * K + k0, ldsA, 64u, 64u, (unsigned)K,
                  (unsigned)K, (unsigned)M);
      tdm_load_2d(Wt + (size_t)k0 * N + bn, ldsB, 64u, 64u, (unsigned)N,
                  (unsigned)N, (unsigned)K);
      __builtin_amdgcn_s_wait_tensorcnt(0);
    }
#else
    *(int4*)(sA + lr * 64 + lc) =
        *(const int4*)(A + (size_t)(bm + lr) * K + k0 + lc);
    *(int4*)(sA + (lr + 32) * 64 + lc) =
        *(const int4*)(A + (size_t)(bm + lr + 32) * K + k0 + lc);
    *(int4*)(sB + lr * 64 + lc) =
        *(const int4*)(Wt + (size_t)(k0 + lr) * N + bn + lc);
    *(int4*)(sB + (lr + 32) * 64 + lc) =
        *(const int4*)(Wt + (size_t)(k0 + lr + 32) * N + bn + lc);
#endif
    __syncthreads();
    U8I a;
    const int abase = (wave * 16 + ln) * 64 + lhi * 8;
#pragma unroll
    for (int j = 0; j < 4; j++)
      a.u64[j] = *(const unsigned long long*)(sA + abase + 16 * j);
#pragma unroll
    for (int t = 0; t < 4; t++) {
      U8I bm8;
      const int nb = t * 16;
#pragma unroll
      for (int s = 0; s < 32; s++)
        ((signed char*)&bm8)[s] =
            sB[(lhi * 8 + 16 * (s >> 3) + (s & 7)) * 64 + nb + ln];
      acc[t].v = __builtin_amdgcn_wmma_i32_16x16x64_iu8(
          true, a.v, true, bm8.v, acc[t].v, false, false);
    }
    __syncthreads();
  }

  const float gw = gwp[0];
#pragma unroll
  for (int t = 0; t < 4; t++) {
#pragma unroll
    for (int r = 0; r < 8; r++) {
      const int m = bm + wave * 16 + r + 8 * lhi;
      const int n = bn + t * 16 + ln;
      float val = (float)acc[t].i32[r] * xs[m] * gw + bias[n];
      if (MODE == 0) {
        val += resid[(size_t)m * N + n];
        outf[(size_t)m * N + n] = val;
      } else if (MODE == 1 || MODE == 2) {
        HU cu; cu.h = (_Float16)val;
        const int h = n >> 6, d = n & 63;
        const int bb = m / SEQ_T, tt = m % SEQ_T;
        size_t idx;
        if (MODE == 1)
          idx = (((size_t)(bb * (N >> 6) + h)) * SEQ_T + tt) * 64 + d;
        else
          idx = (((size_t)(bb * (N >> 6) + h)) * 64 + d) * (size_t)SEQ_T + tt;
        outh[idx] = cu.u;
      } else if (MODE == 3) {
        val = val / (1.f + expf(-val));
        outf[(size_t)m * N + n] = val;
      } else {
        val *= gate[(size_t)m * N + n];
        outf[(size_t)m * N + n] = val;
      }
    }
  }
}

__global__ __launch_bounds__(32) void k_attn(const unsigned short* __restrict__ Qh,
                                             const unsigned short* __restrict__ Kt,
                                             const unsigned short* __restrict__ Vh,
                                             float* __restrict__ attn) {
  __shared__ __align__(16) unsigned short sP[16 * 32];
  const int lane = threadIdx.x;
  const int lhi = lane >> 4, ln = lane & 15;
  const int q0 = blockIdx.x * 16;
  const int bh = blockIdx.y;
  const _Float16* Q  = (const _Float16*)Qh + (size_t)bh * SEQ_T * 64;
  const _Float16* Kp = (const _Float16*)Kt + (size_t)bh * 64 * SEQ_T;
  const _Float16* Vp = (const _Float16*)Vh + (size_t)bh * SEQ_T * 64;

  U16H aq[2];
#pragma unroll
  for (int c = 0; c < 2; c++) {
    const _Float16* qp = Q + (size_t)(q0 + ln) * 64 + c * 32 + lhi * 8;
    aq[c].u64[0] = *(const unsigned long long*)(qp);
    aq[c].u64[1] = *(const unsigned long long*)(qp + 4);
    aq[c].u64[2] = *(const unsigned long long*)(qp + 16);
    aq[c].u64[3] = *(const unsigned long long*)(qp + 20);
  }

  U8F o[4];
  float mrow[8], srow[8];
#pragma unroll
  for (int t = 0; t < 4; t++)
#pragma unroll
    for (int r = 0; r < 8; r++) o[t].f[r] = 0.f;
#pragma unroll
  for (int r = 0; r < 8; r++) { mrow[r] = -3.0e38f; srow[r] = 0.f; }

  const float scale = 0.125f;
  const int ktiles = (q0 + 16 + 31) >> 5;
  for (int kt = 0; kt < ktiles; kt++) {
    const int key0 = kt * 32;
    U8F s[2];
#pragma unroll
    for (int si = 0; si < 2; si++) {
      U16H b0, b1;
      const _Float16* kp0 = Kp + key0 + si * 16 + ln;
#pragma unroll
      for (int s2 = 0; s2 < 16; s2++) {
        const int kk = (s2 < 8) ? (lhi * 8 + s2) : (16 + lhi * 8 + (s2 - 8));
        b0.v[s2] = kp0[(size_t)kk * SEQ_T];
        b1.v[s2] = kp0[(size_t)(kk + 32) * SEQ_T];
      }
      v8f z = {0.f, 0.f, 0.f, 0.f, 0.f, 0.f, 0.f, 0.f};
      z = __builtin_amdgcn_wmma_f32_16x16x32_f16(false, aq[0].v, false, b0.v,
                                                 (short)0, z, false, false);
      z = __builtin_amdgcn_wmma_f32_16x16x32_f16(false, aq[1].v, false, b1.v,
                                                 (short)0, z, false, false);
      s[si].v = z;
    }
#pragma unroll
    for (int r = 0; r < 8; r++) {
      const int qrow = q0 + r + 8 * lhi;
#pragma unroll
      for (int si = 0; si < 2; si++) {
        const int key = key0 + si * 16 + ln;
        float xv = s[si].f[r] * scale;
        if (key > qrow) xv = -1.0e30f;
        s[si].f[r] = xv;
      }
      float mx = fmaxf(s[0].f[r], s[1].f[r]);
      for (int off = 1; off < 16; off <<= 1) mx = fmaxf(mx, __shfl_xor(mx, off, 16));
      const float newm = fmaxf(mrow[r], mx);
      const float corr = expf(mrow[r] - newm);
      const float p0 = expf(s[0].f[r] - newm);
      const float p1 = expf(s[1].f[r] - newm);
      float ts = p0 + p1;
      for (int off = 1; off < 16; off <<= 1) ts += __shfl_xor(ts, off, 16);
      srow[r] = srow[r] * corr + ts;
      mrow[r] = newm;
#pragma unroll
      for (int t = 0; t < 4; t++) o[t].f[r] *= corr;
      s[0].f[r] = p0; s[1].f[r] = p1;
    }
    __syncthreads();
#pragma unroll
    for (int r = 0; r < 8; r++) {
      const int m = r + 8 * lhi;
      HU c0, c1;
      c0.h = (_Float16)s[0].f[r];
      c1.h = (_Float16)s[1].f[r];
      sP[m * 32 + ln]      = c0.u;
      sP[m * 32 + 16 + ln] = c1.u;
    }
    __syncthreads();
    U16H pa;
    pa.u64[0] = *(const unsigned long long*)(sP + ln * 32 + lhi * 8);
    pa.u64[1] = *(const unsigned long long*)(sP + ln * 32 + lhi * 8 + 4);
    pa.u64[2] = *(const unsigned long long*)(sP + ln * 32 + lhi * 8 + 16);
    pa.u64[3] = *(const unsigned long long*)(sP + ln * 32 + lhi * 8 + 20);
#pragma unroll
    for (int t = 0; t < 4; t++) {
      U16H bv;
      const _Float16* vp = Vp + (size_t)key0 * 64 + t * 16 + ln;
#pragma unroll
      for (int s2 = 0; s2 < 16; s2++) {
        const int kk = (s2 < 8) ? (lhi * 8 + s2) : (16 + lhi * 8 + (s2 - 8));
        bv.v[s2] = vp[(size_t)kk * 64];
      }
      o[t].v = __builtin_amdgcn_wmma_f32_16x16x32_f16(false, pa.v, false, bv.v,
                                                      (short)0, o[t].v, false, false);
    }
  }
  const int b = bh >> 4, h = bh & 15;
#pragma unroll
  for (int t = 0; t < 4; t++) {
#pragma unroll
    for (int r = 0; r < 8; r++) {
      const int qrow = q0 + r + 8 * lhi;
      const int tok = b * SEQ_T + qrow;
      const int col = h * 64 + t * 16 + ln;
      attn[(size_t)tok * D_MODEL + col] = o[t].f[r] / srow[r];
    }
  }
}

extern "C" void kernel_launch(void* const* d_in, const int* in_sizes, int n_in,
                              void* d_out, int out_size, void* d_ws, size_t ws_size,
                              hipStream_t stream) {
  (void)in_sizes; (void)n_in; (void)out_size; (void)ws_size;
  const float* x     = (const float*)d_in[0];
  const float* ln1g  = (const float*)d_in[1];
  const float* ln1b  = (const float*)d_in[2];
  const float* wq    = (const float*)d_in[3];
  const float* bq    = (const float*)d_in[4];
  const float* wk    = (const float*)d_in[5];
  const float* bk    = (const float*)d_in[6];
  const float* wv    = (const float*)d_in[7];
  const float* bv    = (const float*)d_in[8];
  const float* wo    = (const float*)d_in[9];
  const float* bo    = (const float*)d_in[10];
  const float* ln2g  = (const float*)d_in[11];
  const float* ln2b  = (const float*)d_in[12];
  const float* wgate = (const float*)d_in[13];
  const float* bgate = (const float*)d_in[14];
  const float* wval  = (const float*)d_in[15];
  const float* bval  = (const float*)d_in[16];
  const float* wout  = (const float*)d_in[17];
  const float* bout  = (const float*)d_in[18];
  float* outp = (float*)d_out;

  char* ws = (char*)d_ws; size_t off = 0;
  auto alloc = [&](size_t bytes) -> char* {
    char* p = ws + off; off += (bytes + 255) & ~(size_t)255; return p;
  };
  float* partial = (float*)alloc(256 * sizeof(float));
  float* gws     = (float*)alloc(8 * sizeof(float));
  signed char* qw[7];
  qw[0] = (signed char*)alloc((size_t)D_MODEL * D_MODEL);
  qw[1] = (signed char*)alloc((size_t)D_MODEL * D_MODEL);
  qw[2] = (signed char*)alloc((size_t)D_MODEL * D_MODEL);
  qw[3] = (signed char*)alloc((size_t)D_MODEL * D_MODEL);
  qw[4] = (signed char*)alloc((size_t)HIDDEN * D_MODEL);
  qw[5] = (signed char*)alloc((size_t)HIDDEN * D_MODEL);
  qw[6] = (signed char*)alloc((size_t)D_MODEL * HIDDEN);
  signed char* xq  = (signed char*)alloc((size_t)NTOK * D_MODEL);
  float* xs1 = (float*)alloc(NTOK * sizeof(float));
  unsigned short* Qhb = (unsigned short*)alloc((size_t)NTOK * D_MODEL * 2);
  unsigned short* Ktb = (unsigned short*)alloc((size_t)NTOK * D_MODEL * 2);
  unsigned short* Vhb = (unsigned short*)alloc((size_t)NTOK * D_MODEL * 2);
  float* attnb = (float*)alloc((size_t)NTOK * D_MODEL * 4);
  signed char* aq1 = (signed char*)alloc((size_t)NTOK * D_MODEL);
  float* as1 = (float*)alloc(NTOK * sizeof(float));
  float* x1  = (float*)alloc((size_t)NTOK * D_MODEL * 4);
  signed char* h2q = (signed char*)alloc((size_t)NTOK * D_MODEL);
  float* h2s = (float*)alloc(NTOK * sizeof(float));
  float* gateb = (float*)alloc((size_t)NTOK * HIDDEN * 4);
  signed char* yq = (signed char*)alloc((size_t)NTOK * HIDDEN);
  float* ys = (float*)alloc(NTOK * sizeof(float));

  struct WI { const float* w; int N; int K; signed char* q; };
  const WI wi[7] = {
      {wq, D_MODEL, D_MODEL, qw[0]},  {wk, D_MODEL, D_MODEL, qw[1]},
      {wv, D_MODEL, D_MODEL, qw[2]},  {wo, D_MODEL, D_MODEL, qw[3]},
      {wgate, HIDDEN, D_MODEL, qw[4]}, {wval, HIDDEN, D_MODEL, qw[5]},
      {wout, D_MODEL, HIDDEN, qw[6]}};
  for (int i = 0; i < 7; i++) {
    const int n = wi[i].N * wi[i].K;
    k_abs_part<<<256, 256, 0, stream>>>(wi[i].w, n, partial);
    k_abs_fin<<<1, 256, 0, stream>>>(partial, n, gws + i);
    k_quant_w_t<<<2048, 256, 0, stream>>>(wi[i].w, gws + i, wi[i].q, wi[i].N, wi[i].K);
  }

  const dim3 gQ(D_MODEL / 64, NTOK / 64);
  const dim3 gF(HIDDEN / 64, NTOK / 64);

  k_ln_quant<<<NTOK, 256, 0, stream>>>(x, ln1g, ln1b, xq, xs1);
  k_gemm_iu8<1><<<gQ, 128, 0, stream>>>(xq, qw[0], xs1, gws + 0, bq, nullptr,
                                        nullptr, Qhb, nullptr, NTOK, D_MODEL, D_MODEL);
  k_gemm_iu8<2><<<gQ, 128, 0, stream>>>(xq, qw[1], xs1, gws + 1, bk, nullptr,
                                        nullptr, Ktb, nullptr, NTOK, D_MODEL, D_MODEL);
  k_gemm_iu8<1><<<gQ, 128, 0, stream>>>(xq, qw[2], xs1, gws + 2, bv, nullptr,
                                        nullptr, Vhb, nullptr, NTOK, D_MODEL, D_MODEL);
  k_attn<<<dim3(SEQ_T / 16, 2 * NHEADS), 32, 0, stream>>>(Qhb, Ktb, Vhb, attnb);
  k_quant_rows<<<NTOK, 256, 0, stream>>>(attnb, aq1, as1, D_MODEL);
  k_gemm_iu8<0><<<gQ, 128, 0, stream>>>(aq1, qw[3], as1, gws + 3, bo, x,
                                        x1, nullptr, nullptr, NTOK, D_MODEL, D_MODEL);

  k_ln_quant<<<NTOK, 256, 0, stream>>>(x1, ln2g, ln2b, h2q, h2s);
  k_gemm_iu8<3><<<gF, 128, 0, stream>>>(h2q, qw[4], h2s, gws + 4, bgate, nullptr,
                                        gateb, nullptr, nullptr, NTOK, HIDDEN, D_MODEL);
  k_gemm_iu8<4><<<gF, 128, 0, stream>>>(h2q, qw[5], h2s, gws + 5, bval, nullptr,
                                        gateb, nullptr, gateb, NTOK, HIDDEN, D_MODEL);
  k_quant_rows<<<NTOK, 256, 0, stream>>>(gateb, yq, ys, HIDDEN);
  k_gemm_iu8<0><<<gQ, 128, 0, stream>>>(yq, qw[6], ys, gws + 6, bout, x1,
                                        outp, nullptr, nullptr, NTOK, D_MODEL, HIDDEN);
}
